// MoELayer_13563506721407
// MI455X (gfx1250) — compile-verified, matrix operand order checked by reading, not run on hardware
//
#include <hip/hip_runtime.h>

#define DD   512
#define HH   2048
#define EE   8
#define NTOK 4096
#define TM   16      // token tile (M)
#define HCH  1024    // H chunk held in LDS as bf16

typedef __attribute__((ext_vector_type(16))) __bf16 v16bf;
typedef __attribute__((ext_vector_type(8)))  float  v8f;

// WMMA 16-bit operand layout (ISA 7.12.2, 16x32 A-matrix):
//   register slot i (0..15), half-wave hi (0/1)  ->  K index in the 32-wide K tile
__device__ __forceinline__ int kfwd(int i, int hi) {
    return i + hi * 8 + ((i >= 8) ? 8 : 0);
}
// inverse: K index kk (0..31) -> (hi, slot)
__device__ __forceinline__ void kinv(int kk, int& hi, int& slot) {
    hi   = (kk >> 3) & 1;
    slot = kk - hi * 8 - ((kk >= 16) ? 8 : 0);
}

__global__ void init_counts(int* counts) {
    if (threadIdx.x < EE) counts[threadIdx.x] = 0;
}

__global__ void router_kernel(const float* __restrict__ x,
                              const float* __restrict__ Wg,
                              const float* __restrict__ bg,
                              int* __restrict__ counts,
                              int* __restrict__ lists,
                              float* __restrict__ scores) {
    int t = blockIdx.x * blockDim.x + threadIdx.x;
    if (t >= NTOK) return;
    float l[EE];
#pragma unroll
    for (int e = 0; e < EE; ++e) l[e] = bg[e];
    const float* xr = x + (size_t)t * DD;
    for (int d = 0; d < DD; ++d) {
        float xv = xr[d];
#pragma unroll
        for (int e = 0; e < EE; ++e) l[e] = fmaf(xv, Wg[d * EE + e], l[e]);
    }
    int best = 0; float mx = l[0];
#pragma unroll
    for (int e = 1; e < EE; ++e) { if (l[e] > mx) { mx = l[e]; best = e; } }
    float s = 0.f;
#pragma unroll
    for (int e = 0; e < EE; ++e) s += __expf(l[e] - mx);
    scores[t] = 1.0f / s;                 // top-1 softmax prob = exp(0)/sum
    int pos = atomicAdd(&counts[best], 1);
    lists[best * NTOK + pos] = t;
}

__global__ __launch_bounds__(256)
void expert_kernel(const float* __restrict__ x,
                   const float* __restrict__ W1, const float* __restrict__ b1,
                   const float* __restrict__ W2, const float* __restrict__ b2,
                   const int* __restrict__ counts, const int* __restrict__ lists,
                   const float* __restrict__ scores, float* __restrict__ out) {
    // fragment-major swizzled operand tiles: [ktile][lane][slot], 32B per lane-row
    __shared__ __align__(32) __bf16 s_x[DD / 32][32][16];   // 16 KB
    __shared__ __align__(32) __bf16 s_h[HCH / 32][32][16];  // 32 KB
    __shared__ int   s_tok[TM];
    __shared__ float s_sc[TM];

    const int e    = blockIdx.x;
    const int tile = blockIdx.y;
    const int cnt  = counts[e];
    const int base = tile * TM;
    if (base >= cnt) return;
    const int nvalid = min(TM, cnt - base);

    const int tid    = threadIdx.x;
    const int wave   = tid >> 5;
    const int lane   = tid & 31;
    const int laneHi = lane >> 4;
    const int ln     = lane & 15;

    if (tid < TM) {
        int r   = min(tid, nvalid - 1);
        int tok = lists[e * NTOK + base + r];
        s_tok[tid] = tok;
        s_sc[tid]  = scores[tok];
    }
    __syncthreads();

    // stage X tile (f32 -> bf16), swizzled into fragment-major order
    for (int j = tid; j < TM * DD; j += 256) {
        int m = j >> 9;            // /DD
        int c = j & (DD - 1);
        int hi, slot; kinv(c & 31, hi, slot);
        s_x[c >> 5][hi * 16 + m][slot] = (__bf16)x[(size_t)s_tok[m] * DD + c];
    }

    const float* W1e = W1 + (size_t)e * DD * HH;
    const float* W2e = W2 + (size_t)e * HH * DD;

    // per-wave output accumulators: 4 statically-named N-tiles of 16 cols
    // (8 waves * 64 = D); static names keep them pinned in WMMA acc VGPRs.
    v8f acc2_0, acc2_1, acc2_2, acc2_3;
    {
        const float* b2e = b2 + (size_t)e * DD + wave * 64 + ln;
        float b0 = b2e[0], b1v = b2e[16], b2v = b2e[32], b3 = b2e[48];
#pragma unroll
        for (int r = 0; r < 8; ++r) {
            acc2_0[r] = b0; acc2_1[r] = b1v; acc2_2[r] = b2v; acc2_3[r] = b3;
        }
    }
    __syncthreads();

    for (int ch = 0; ch < HH / HCH; ++ch) {
        const int hbase = ch * HCH;

        // ---- phase 1: h[:, hbase:hbase+HCH] = relu(X @ W1 + b1) ----
#pragma unroll 1
        for (int nt = 0; nt < HCH / (16 * 8); ++nt) {   // 8 N-tiles per wave
            const int n0 = hbase + (wave * 8 + nt) * 16;
            v8f acc;
            float bv = b1[(size_t)e * HH + n0 + ln];
#pragma unroll
            for (int r = 0; r < 8; ++r) acc[r] = bv;
#pragma unroll 1
            for (int k0 = 0; k0 < DD; k0 += 32) {
                v16bf a = *reinterpret_cast<const v16bf*>(&s_x[k0 >> 5][lane][0]);
                v16bf b;
#pragma unroll
                for (int i = 0; i < 16; ++i) {
                    int k = kfwd(i, laneHi);
                    b[i] = (__bf16)W1e[(size_t)(k0 + k) * HH + n0 + ln];
                }
                if (k0 + 32 < DD)
                    __builtin_prefetch(&W1e[(size_t)(k0 + 32 + laneHi * 8) * HH + n0 + ln], 0, 1);
                acc = __builtin_amdgcn_wmma_f32_16x16x32_bf16(
                          false, a, false, b, (short)0, acc, false, false);
            }
            // relu + swizzled bf16 store (phase-2 operand layout)
            {
                int cl = (n0 - hbase) + ln;      // column within chunk == K for phase 2
                int hi, slot; kinv(cl & 31, hi, slot);
#pragma unroll
                for (int r = 0; r < 8; ++r) {
                    int m = r + laneHi * 8;
                    float v = acc[r];
                    v = v > 0.f ? v : 0.f;
                    s_h[cl >> 5][hi * 16 + m][slot] = (__bf16)v;
                }
            }
        }
        __syncthreads();

        // ---- phase 2 (partial): out += h_chunk @ W2[hbase:hbase+HCH, :] ----
        // fully unrolled over the 4 N-tiles so accumulators stay statically named
#define PHASE2_NTILE(ACC, NT)                                                        \
        {                                                                            \
            const int n0 = (wave * 4 + (NT)) * 16;                                   \
            v8f acc = ACC;                                                           \
            _Pragma("unroll 1")                                                      \
            for (int k0 = 0; k0 < HCH; k0 += 32) {                                   \
                v16bf a = *reinterpret_cast<const v16bf*>(&s_h[k0 >> 5][lane][0]);   \
                v16bf b;                                                             \
                _Pragma("unroll")                                                    \
                for (int i = 0; i < 16; ++i) {                                       \
                    int k = kfwd(i, laneHi);                                         \
                    b[i] = (__bf16)W2e[(size_t)(hbase + k0 + k) * DD + n0 + ln];     \
                }                                                                    \
                if (k0 + 32 < HCH)                                                   \
                    __builtin_prefetch(                                              \
                        &W2e[(size_t)(hbase + k0 + 32 + laneHi * 8) * DD + n0 + ln], \
                        0, 1);                                                       \
                acc = __builtin_amdgcn_wmma_f32_16x16x32_bf16(                       \
                          false, a, false, b, (short)0, acc, false, false);          \
            }                                                                        \
            ACC = acc;                                                               \
        }
        PHASE2_NTILE(acc2_0, 0)
        PHASE2_NTILE(acc2_1, 1)
        PHASE2_NTILE(acc2_2, 2)
        PHASE2_NTILE(acc2_3, 3)
#undef PHASE2_NTILE

        __syncthreads();    // before next chunk overwrites s_h
    }

    // ---- finalize: scale by routing score, scatter to out rows ----
#define STORE_NTILE(ACC, NT)                                                         \
    {                                                                                \
        const int n0 = (wave * 4 + (NT)) * 16;                                       \
        _Pragma("unroll")                                                            \
        for (int r = 0; r < 8; ++r) {                                                \
            int m = r + laneHi * 8;                                                  \
            if (m < nvalid)                                                          \
                out[(size_t)s_tok[m] * DD + n0 + ln] = ACC[r] * s_sc[m];             \
        }                                                                            \
    }
    STORE_NTILE(acc2_0, 0)
    STORE_NTILE(acc2_1, 1)
    STORE_NTILE(acc2_2, 2)
    STORE_NTILE(acc2_3, 3)
#undef STORE_NTILE
}

extern "C" void kernel_launch(void* const* d_in, const int* in_sizes, int n_in,
                              void* d_out, int out_size, void* d_ws, size_t ws_size,
                              hipStream_t stream) {
    const float* x  = (const float*)d_in[0];
    const float* Wg = (const float*)d_in[1];
    const float* bg = (const float*)d_in[2];
    const float* W1 = (const float*)d_in[3];
    const float* b1 = (const float*)d_in[4];
    const float* W2 = (const float*)d_in[5];
    const float* b2 = (const float*)d_in[6];
    float* out = (float*)d_out;

    int*   counts = (int*)d_ws;
    int*   lists  = counts + 16;                // 8 counters, padded
    float* scores = (float*)(lists + EE * NTOK);

    init_counts<<<1, 64, 0, stream>>>(counts);
    router_kernel<<<NTOK / 256, 256, 0, stream>>>(x, Wg, bg, counts, lists, scores);
    expert_kernel<<<dim3(EE, NTOK / TM), 256, 0, stream>>>(
        x, W1, b1, W2, b2, counts, lists, scores, out);
}
